// MultiHeadAttention_36721970380800
// MI455X (gfx1250) — compile-verified, matrix operand order checked by reading, not run on hardware
//
#include <hip/hip_runtime.h>
#include <hip/hip_bf16.h>

typedef __attribute__((ext_vector_type(16))) _Float16 v16h;
typedef __attribute__((ext_vector_type(8)))  _Float16 v8h;
typedef __attribute__((ext_vector_type(8)))  float    v8f;

#define B_SZ   4
#define S_LEN  2048
#define NHEAD  8
#define DHEAD  64
#define DIM    512

// 1/sqrt(DHEAD) * log2(e): folds softmax scaling AND the exp->exp2 change of
// base into the Q projection (applied in fp32 there).
#define Q_SCALE (0.125f * 1.4426950408889634f)

// Build a 16-bit WMMA A/B operand for 16x16x32:
// lane (g = lane>>4) holds halves K = {8g..8g+7} and {16+8g..16+8g+7},
// i.e. two contiguous 16-byte runs from a K-major row/column.
static __device__ __forceinline__ v16h load2x8(const _Float16* base, int g) {
  const v8h lo = *(const v8h*)(base + 8 * g);
  const v8h hi = *(const v8h*)(base + 16 + 8 * g);
  v16h r;
#pragma unroll
  for (int i = 0; i < 8; ++i) { r[i] = lo[i]; r[i + 8] = hi[i]; }
  return r;
}

// ---------------------------------------------------------------------------
// Kernel 1: Y = (X @ W + b) * oscale, converted to f16, written head-split.
//   transposed==0 : dst[((b*H + h)*S + s)*64 + d]   (Q, K)
//   transposed==1 : dst[((b*H + h)*64 + d)*S + s]   (V^T for PV B-operand)
// Block = 128 threads (4 waves); block tile = 64 rows x 64 cols; K step 32.
// ---------------------------------------------------------------------------
__global__ __launch_bounds__(128) void qkv_proj_kernel(
    const float* __restrict__ X, const float* __restrict__ W,
    const float* __restrict__ bias, _Float16* __restrict__ dst,
    int transposed, float oscale) {
  __shared__ __align__(16) _Float16 ldsW[64 * 32];  // [col][k] transposed tile

  const int tid  = threadIdx.x;
  const int w    = tid >> 5;
  const int lane = tid & 31;
  const int g    = lane >> 4;
  const int ln   = lane & 15;
  const int rbase = blockIdx.y * 64;
  const int cbase = blockIdx.x * 64;

  v8f acc[4] = {};
  const int    arow = rbase + 16 * w + ln;    // A operand: lane = row
  const float* xrow = X + (long)arow * DIM;

  for (int kb = 0; kb < DIM / 32; ++kb) {
    __syncthreads();
    // Stage W[kb*32 .. +32)[cbase .. +64) into LDS as f16, [col][k].
#pragma unroll
    for (int i = 0; i < 16; ++i) {
      int linear = tid + 128 * i;          // 0..2047
      int cc = linear & 63;
      int kk = linear >> 6;
      ldsW[cc * 32 + kk] = (_Float16)W[(long)(kb * 32 + kk) * DIM + cbase + cc];
    }
    __syncthreads();

    // A operand (convert f32 -> f16 on the fly)
    const float* ab = xrow + kb * 32;
    v16h A;
#pragma unroll
    for (int j = 0; j < 8; ++j) {
      A[j]     = (_Float16)ab[8 * g + j];
      A[j + 8] = (_Float16)ab[16 + 8 * g + j];
    }

#pragma unroll
    for (int t = 0; t < 4; ++t) {
      v16h B = load2x8(&ldsW[(16 * t + ln) * 32], g);
      acc[t] = __builtin_amdgcn_wmma_f32_16x16x32_f16(
          false, A, false, B, (short)0, acc[t], false, false);
    }
  }

  // Epilogue: bias add, output scale (fp32), f16 convert, head-split store.
#pragma unroll
  for (int t = 0; t < 4; ++t) {
    const int   c  = cbase + 16 * t + ln;
    const float bv = bias[c];
    const int   h  = c >> 6;
    const int   d  = c & 63;
#pragma unroll
    for (int j = 0; j < 8; ++j) {
      int rg = rbase + 16 * w + 8 * g + j;  // C layout: row = 8g + j
      int b  = rg >> 11;                    // S = 2048
      int s  = rg & (S_LEN - 1);
      long idx = transposed
                     ? (((long)(b * NHEAD + h) * DHEAD + d) * S_LEN + s)
                     : (((long)(b * NHEAD + h) * S_LEN + s) * DHEAD + d);
      dst[idx] = (_Float16)((acc[t][j] + bv) * oscale);
    }
  }
}

// ---------------------------------------------------------------------------
// Kernel 2: flash attention, log2-domain softmax. Grid (S/64, H, B),
// 128 threads = 4 waves. Wave w owns q-rows [qb*64 + 16w, +16) x Dh=64
// (4 fp32 accumulators); sweeps 32 key blocks of 64.
// Q was pre-scaled by 1/sqrt(Dh)*log2(e), so p = exp2(s - m) directly
// (single v_exp_f32 per score). Row-sum reduction is deferred: each lane
// carries a partial l over its 4 columns; one butterfly in the epilogue.
// ---------------------------------------------------------------------------
__global__ __launch_bounds__(128) void attn_kernel(
    const _Float16* __restrict__ Q, const _Float16* __restrict__ K,
    const _Float16* __restrict__ VT, float* __restrict__ out) {
  __shared__ __align__(16) _Float16 ldsP[4 * 16 * DHEAD];  // wave-private slabs

  const int tid  = threadIdx.x;
  const int w    = tid >> 5;
  const int lane = tid & 31;
  const int g    = lane >> 4;
  const int ln   = lane & 15;
  const int qb = blockIdx.x, h = blockIdx.y, b = blockIdx.z;

  const long bh = (long)b * NHEAD + h;
  const _Float16* qp = Q  + bh * S_LEN * DHEAD;
  const _Float16* kp = K  + bh * S_LEN * DHEAD;
  const _Float16* vp = VT + bh * DHEAD * S_LEN;

  // Q operand, loop-invariant (already carries softmax scale in log2 units).
  const int qrow = qb * 64 + 16 * w + ln;
  v16h Aq[2];
#pragma unroll
  for (int kc = 0; kc < 2; ++kc)
    Aq[kc] = load2x8(qp + (long)qrow * DHEAD + kc * 32, g);

  float m[8], l[8];
  v8f acc[4] = {};
#pragma unroll
  for (int j = 0; j < 8; ++j) { m[j] = -1e30f; l[j] = 0.0f; }

  _Float16* myP = &ldsP[w * 16 * DHEAD];

  for (int kb = 0; kb < S_LEN / 64; ++kb) {
    // ---- scores (log2 units): S = Qs @ K^T, 16 x 64 per wave ----
    v8f sc[4];
#pragma unroll
    for (int t = 0; t < 4; ++t) {
      const _Float16* kr = kp + (long)(kb * 64 + 16 * t + ln) * DHEAD;
      v16h B0 = load2x8(kr, g);
      v16h B1 = load2x8(kr + 32, g);
      v8f z = {};
      z = __builtin_amdgcn_wmma_f32_16x16x32_f16(false, Aq[0], false, B0,
                                                 (short)0, z, false, false);
      sc[t] = __builtin_amdgcn_wmma_f32_16x16x32_f16(false, Aq[1], false, B1,
                                                     (short)0, z, false, false);
    }

    // ---- row max (butterfly over the 16-lane C-layout group) ----
    float mn[8];
#pragma unroll
    for (int j = 0; j < 8; ++j) {
      float v = fmaxf(fmaxf(sc[0][j], sc[1][j]), fmaxf(sc[2][j], sc[3][j]));
#pragma unroll
      for (int msk = 1; msk < 16; msk <<= 1) v = fmaxf(v, __shfl_xor(v, msk, 32));
      mn[j] = fmaxf(m[j], v);
    }

    // ---- p = 2^(s - m); lane-partial row sums; stash P as f16 in LDS ----
    float corr[8];
#pragma unroll
    for (int j = 0; j < 8; ++j) {
      corr[j] = __builtin_amdgcn_exp2f(m[j] - mn[j]);
      m[j] = mn[j];
      l[j] *= corr[j];
    }
#pragma unroll
    for (int t = 0; t < 4; ++t) {
#pragma unroll
      for (int j = 0; j < 8; ++j) {
        float p = __builtin_amdgcn_exp2f(sc[t][j] - mn[j]);
        l[j] += p;  // partial: this lane's 4 columns only
        myP[(8 * g + j) * DHEAD + 16 * t + ln] = (_Float16)p;
      }
      acc[t] *= *(const v8f*)corr;  // rescale output accumulator
    }

    // ---- PV: acc += P @ V (V stored transposed -> contiguous B loads) ----
    v16h Ap0 = load2x8(myP + ln * DHEAD, g);
    v16h Ap1 = load2x8(myP + ln * DHEAD + 32, g);
#pragma unroll
    for (int t = 0; t < 4; ++t) {
      const _Float16* vb = vp + (long)(16 * t + ln) * S_LEN + kb * 64;
      v16h Bv0 = load2x8(vb, g);
      v16h Bv1 = load2x8(vb + 32, g);
      acc[t] = __builtin_amdgcn_wmma_f32_16x16x32_f16(false, Ap0, false, Bv0,
                                                      (short)0, acc[t], false, false);
      acc[t] = __builtin_amdgcn_wmma_f32_16x16x32_f16(false, Ap1, false, Bv1,
                                                      (short)0, acc[t], false, false);
    }
  }

  // ---- epilogue: finish deferred row-sum, normalize, store fp32 ----
#pragma unroll
  for (int j = 0; j < 8; ++j) {
    float v = l[j];
#pragma unroll
    for (int msk = 1; msk < 16; msk <<= 1) v += __shfl_xor(v, msk, 32);
    l[j] = 1.0f / v;
  }
#pragma unroll
  for (int t = 0; t < 4; ++t) {
#pragma unroll
    for (int j = 0; j < 8; ++j) {
      int s = qb * 64 + 16 * w + 8 * g + j;
      out[((long)b * S_LEN + s) * DIM + h * DHEAD + 16 * t + ln] =
          acc[t][j] * l[j];
    }
  }
}

// ---------------------------------------------------------------------------
extern "C" void kernel_launch(void* const* d_in, const int* in_sizes, int n_in,
                              void* d_out, int out_size, void* d_ws, size_t ws_size,
                              hipStream_t stream) {
  const float* q  = (const float*)d_in[0];
  const float* k  = (const float*)d_in[1];
  const float* v  = (const float*)d_in[2];
  const float* Wq = (const float*)d_in[3];
  const float* bq = (const float*)d_in[4];
  const float* Wk = (const float*)d_in[5];
  const float* bk = (const float*)d_in[6];
  const float* Wv = (const float*)d_in[7];
  const float* bv = (const float*)d_in[8];

  const size_t per = (size_t)B_SZ * NHEAD * S_LEN * DHEAD;  // f16 elements
  _Float16* wsq = (_Float16*)d_ws;
  _Float16* wsk = wsq + per;
  _Float16* wsv = wsk + per;

  dim3 blk(128);
  dim3 g1(DIM / 64, (B_SZ * S_LEN) / 64);   // (8, 128)
  qkv_proj_kernel<<<g1, blk, 0, stream>>>(q, Wq, bq, wsq, 0, Q_SCALE);
  qkv_proj_kernel<<<g1, blk, 0, stream>>>(k, Wk, bk, wsk, 0, 1.0f);
  qkv_proj_kernel<<<g1, blk, 0, stream>>>(v, Wv, bv, wsv, 1, 1.0f);

  dim3 g2(S_LEN / 64, NHEAD, B_SZ);         // (32, 8, 4)
  attn_kernel<<<g2, blk, 0, stream>>>(wsq, wsk, wsv, (float*)d_out);
}
